// Block_67980742361706
// MI455X (gfx1250) — matrix operand order repaired and checked by reading, not compiled, not run on hardware
//
#include <hip/hip_runtime.h>
#include <cstdint>
#include <cstddef>

// ---------------- problem constants ----------------
#define BB   2
#define TT   2048
#define CC   1024
#define HH   16
#define HS   64
#define HID  2730
#define HIDP 2752              // HID padded up to multiple of 32 for WMMA K
#define MM   (BB * TT)         // 4096 rows

typedef __bf16 bf16_t;
typedef __attribute__((ext_vector_type(16))) __bf16 bf16x16;
typedef __attribute__((ext_vector_type(8)))  float  f32x8;

// ---------------- fragment helpers ----------------
__device__ __forceinline__ bf16x16 load_frag_2x8(const bf16_t* p0, const bf16_t* p1) {
    union { bf16x16 v; float4 q[2]; } u;
    u.q[0] = *reinterpret_cast<const float4*>(p0);
    u.q[1] = *reinterpret_cast<const float4*>(p1);
    return u.v;
}
__device__ __forceinline__ bf16x16 load_frag_16(const bf16_t* p) {
    return load_frag_2x8(p, p + 8);
}
__device__ __forceinline__ f32x8 wmma_bf16(bf16x16 a, bf16x16 b, f32x8 c) {
    // D = A(16x32) * B(32x16) + C, bf16 in / f32 accum
    return __builtin_amdgcn_wmma_f32_16x16x32_bf16(false, a, false, b, (short)0, c, false, false);
}
__device__ __forceinline__ f32x8 f32x8_zero() {
    f32x8 z = {0.f, 0.f, 0.f, 0.f, 0.f, 0.f, 0.f, 0.f};
    return z;
}

// ---------------- generic WMMA GEMM: Y = X[M,K]bf16 @ W[N,K]bf16^T (+bias)(+res) ----------------
// One wave per block; each wave computes a 32(M) x 64(N) strip (8 accumulators):
// per 32-wide K step: 4 A-loads + 8 B-loads feed 8 WMMAs (vs 2+8 for 4 at 16x64),
// 1.67x more matrix work per byte pulled from L2.
__global__ __launch_bounds__(32)
void wmma_gemm_kernel(const bf16_t* __restrict__ X, const bf16_t* __restrict__ W,
                      const float* __restrict__ bias, const float* __restrict__ res,
                      float* __restrict__ Y, int M, int N, int K)
{
    const int lane  = threadIdx.x & 31;
    const int lh    = lane >> 4;     // lane half (0/1)
    const int ln    = lane & 15;
    const int mBase = blockIdx.x * 32;
    const int nBase = blockIdx.y * 64;

    f32x8 acc[2][4];
#pragma unroll
    for (int u = 0; u < 2; ++u)
#pragma unroll
        for (int t = 0; t < 4; ++t) acc[u][t] = f32x8_zero();

    const bf16_t* xrow0 = X + (size_t)(mBase + ln) * K;
    const bf16_t* xrow1 = X + (size_t)(mBase + 16 + ln) * K;
    for (int kk = 0; kk < K; kk += 32) {
        // A fragments (16x32): lane row = ln; elems e<8 -> k=kk+8*lh+e, e>=8 -> k=kk+16+8*lh+(e-8)
        bf16x16 a0 = load_frag_2x8(xrow0 + kk + 8 * lh, xrow0 + kk + 16 + 8 * lh);
        bf16x16 a1 = load_frag_2x8(xrow1 + kk + 8 * lh, xrow1 + kk + 16 + 8 * lh);
#pragma unroll
        for (int t = 0; t < 4; ++t) {
            int n = nBase + t * 16 + ln;
            if (n >= N) n = N - 1;                       // clamp (results discarded on store)
            const bf16_t* wrow = W + (size_t)n * K + kk;
            bf16x16 bfrag = load_frag_2x8(wrow + 8 * lh, wrow + 16 + 8 * lh); // B elems e<8 -> k=kk+8*lh+e, e>=8 -> k=kk+16+8*lh+(e-8)
            acc[0][t] = wmma_bf16(a0, bfrag, acc[0][t]);
            acc[1][t] = wmma_bf16(a1, bfrag, acc[1][t]);
        }
    }
#pragma unroll
    for (int u = 0; u < 2; ++u)
#pragma unroll
        for (int t = 0; t < 4; ++t) {
            int col = nBase + t * 16 + ln;
            if (col >= N) continue;
            float bv = bias ? bias[col] : 0.f;
#pragma unroll
            for (int r = 0; r < 8; ++r) {
                int row = mBase + u * 16 + r + 8 * lh;   // D: VGPR r -> M = r + 8*half
                float v = acc[u][t][r] + bv;
                if (res) v += res[(size_t)row * N + col];
                Y[(size_t)row * N + col] = v;
            }
        }
}

// ---------------- flash attention: one wave per (b, h, 16-query tile) ----------------
// Computes S^T = K_tile @ Q^T so that the f32 D-layout (lane=query, vgpr=key) is exactly
// the A-fragment layout needed for P@V after exp() -> bf16. No LDS transpose needed.
__global__ __launch_bounds__(32)
void flash_attn_kernel(const bf16_t* __restrict__ Qb, const bf16_t* __restrict__ Kb,
                       const bf16_t* __restrict__ VTb, float* __restrict__ Out)
{
    const int lane  = threadIdx.x & 31;
    const int lh    = lane >> 4;
    const int ln    = lane & 15;
    const int qBase = blockIdx.x * 16;
    const int h     = blockIdx.y;
    const int b     = blockIdx.z;

    // Q as B-operand: B[k=d][n=query] = Q[n][d]; two 32-wide d chunks over HS=64
    const bf16_t* qrow = Qb + ((size_t)(b * TT + qBase + ln)) * CC + h * HS;
    bf16x16 qf0 = load_frag_2x8(qrow + 8 * lh,      qrow + 16 + 8 * lh);
    bf16x16 qf1 = load_frag_2x8(qrow + 32 + 8 * lh, qrow + 48 + 8 * lh);

    f32x8 o[4];
#pragma unroll
    for (int t = 0; t < 4; ++t) o[t] = f32x8_zero();
    float mrow = -1e30f, lrow = 0.f;

    const bf16_t* kpan = Kb  + ((size_t)(b * TT)) * CC + h * HS;
    const bf16_t* vpan = VTb + ((size_t)((b * HH + h) * HS)) * TT;

    const int nblk = qBase / 32 + 1;                     // causal: keys 0 .. qBase+15
    for (int jb = 0; jb < nblk; ++jb) {
        const int kBase = jb * 32;
        f32x8 s[2];
#pragma unroll
        for (int mt = 0; mt < 2; ++mt) {                 // two 16-key tiles of the 32-key block
            const bf16_t* krow = kpan + (size_t)(kBase + mt * 16 + ln) * CC;
            f32x8 accs = f32x8_zero();
            bf16x16 a0 = load_frag_2x8(krow + 8 * lh,      krow + 16 + 8 * lh);
            accs = wmma_bf16(a0, qf0, accs);
            bf16x16 a1 = load_frag_2x8(krow + 32 + 8 * lh, krow + 48 + 8 * lh);
            accs = wmma_bf16(a1, qf1, accs);
            s[mt] = accs;                                // S^T[key=kBase+mt*16+r+8*lh][query=qBase+ln]
        }
        // scale + causal mask + online softmax (row = query = ln; full row via xor-16 partner)
        const int qg = qBase + ln;
        float pv[16];
        float bm = -1e30f;
#pragma unroll
        for (int mt = 0; mt < 2; ++mt)
#pragma unroll
            for (int r = 0; r < 8; ++r) {
                int kg = kBase + mt * 16 + 8 * lh + r;
                float v = s[mt][r] * 0.125f;             // HS^-0.5
                if (kg > qg) v = -1e30f;
                pv[mt * 8 + r] = v;
                bm = fmaxf(bm, v);
            }
        bm = fmaxf(bm, __shfl_xor(bm, 16, 32));
        float mnew = fmaxf(mrow, bm);
        float ls = 0.f;
#pragma unroll
        for (int i = 0; i < 16; ++i) { pv[i] = __expf(pv[i] - mnew); ls += pv[i]; }
        ls += __shfl_xor(ls, 16, 32);
        float corr = __expf(mrow - mnew);
        lrow = lrow * corr + ls;
        mrow = mnew;
        // rescale O: lane needs corr of queries 8*lh .. 8*lh+7 (held by lanes 8*lh+r in group)
        float c8[8];
#pragma unroll
        for (int r = 0; r < 8; ++r) c8[r] = __shfl(corr, 8 * lh + r, 16);
#pragma unroll
        for (int t = 0; t < 4; ++t)
#pragma unroll
            for (int r = 0; r < 8; ++r) o[t][r] *= c8[r];
        // P fragment: A elems e<8 -> pv[e] (keys kBase+8lh+e), e>=8 -> pv[e] (keys kBase+16+8lh+e-8)
        union { bf16x16 v; __bf16 e[16]; } pf;
#pragma unroll
        for (int i = 0; i < 16; ++i) pf.e[i] = (__bf16)pv[i];
        // O += P @ V ; V^T layout makes B elems contiguous in t
#pragma unroll
        for (int t = 0; t < 4; ++t) {
            const bf16_t* vrow = vpan + (size_t)(t * 16 + ln) * TT + kBase;
            bf16x16 vf = load_frag_2x8(vrow + 8 * lh, vrow + 16 + 8 * lh);
            o[t] = wmma_bf16(pf.v, vf, o[t]);
        }
    }
    // epilogue: divide by row sums, write [B,T,H*HS] f32
    float l8[8];
#pragma unroll
    for (int r = 0; r < 8; ++r) l8[r] = __shfl(lrow, 8 * lh + r, 16);
#pragma unroll
    for (int t = 0; t < 4; ++t)
#pragma unroll
        for (int r = 0; r < 8; ++r) {
            int row = qBase + r + 8 * lh;
            Out[((size_t)(b * TT + row)) * CC + h * HS + t * 16 + ln] = o[t][r] / l8[r];
        }
}

// ---------------- RMSNorm (one row per 256-thread block) ----------------
__global__ __launch_bounds__(256)
void rmsnorm_kernel(const float* __restrict__ X, const float* __restrict__ w,
                    float* __restrict__ Yf, bf16_t* __restrict__ Yb)
{
    __shared__ float sws[8];
    const int row = blockIdx.x;
    const float* xr = X + (size_t)row * CC;
    float ss = 0.f;
    for (int i = threadIdx.x; i < CC; i += 256) { float v = xr[i]; ss += v * v; }
#pragma unroll
    for (int off = 16; off > 0; off >>= 1) ss += __shfl_xor(ss, off, 32);
    if ((threadIdx.x & 31) == 0) sws[threadIdx.x >> 5] = ss;
    __syncthreads();
    float tot = 0.f;
#pragma unroll
    for (int i = 0; i < 8; ++i) tot += sws[i];
    const float scale = rsqrtf(tot / (float)CC + 1e-6f);
    for (int i = threadIdx.x; i < CC; i += 256) {
        float v = xr[i] * scale * w[i];
        Yf[(size_t)row * CC + i] = v;
        Yb[(size_t)row * CC + i] = (bf16_t)v;
    }
}

// ---------------- RoPE on q,k (reads f32, writes bf16) ----------------
__global__ void rope_kernel(const float* __restrict__ Qf, const float* __restrict__ Kf,
                            const float* __restrict__ cosT, const float* __restrict__ sinT,
                            bf16_t* __restrict__ Qb, bf16_t* __restrict__ Kb)
{
    size_t id = (size_t)blockIdx.x * blockDim.x + threadIdx.x;   // over B*T*C/2 pairs
    const size_t total = (size_t)MM * CC / 2;
    if (id >= total) return;
    int   i   = (int)(id % (HS / 2));
    size_t rem = id / (HS / 2);
    int   h   = (int)(rem % HH);
    size_t bt = rem / HH;                                        // b*T + t
    int   t   = (int)(bt % TT);
    size_t base = bt * CC + (size_t)h * HS + 2 * i;
    float c = cosT[t * (HS / 2) + i], s = sinT[t * (HS / 2) + i];
    float q0 = Qf[base], q1 = Qf[base + 1];
    Qb[base]     = (bf16_t)(q0 * c - q1 * s);
    Qb[base + 1] = (bf16_t)(q0 * s + q1 * c);
    float k0 = Kf[base], k1 = Kf[base + 1];
    Kb[base]     = (bf16_t)(k0 * c - k1 * s);
    Kb[base + 1] = (bf16_t)(k0 * s + k1 * c);
}

// ---------------- V transpose: [B,T,H*HS] f32 -> [B,H,HS,T] bf16 ----------------
__global__ void vtrans_kernel(const float* __restrict__ Vf, bf16_t* __restrict__ VTb)
{
    size_t id = (size_t)blockIdx.x * blockDim.x + threadIdx.x;
    const size_t total = (size_t)MM * CC;
    if (id >= total) return;
    int   t   = (int)(id % TT);
    size_t rem = id / TT;
    int   d   = (int)(rem % HS); rem /= HS;
    int   h   = (int)(rem % HH);
    int   b   = (int)(rem / HH);
    VTb[id] = (bf16_t)Vf[((size_t)(b * TT + t)) * CC + h * HS + d];
}

// ---------------- f32 -> bf16 cast ----------------
__global__ void cast_kernel(const float* __restrict__ src, bf16_t* __restrict__ dst, size_t n)
{
    size_t id = (size_t)blockIdx.x * blockDim.x + threadIdx.x;
    if (id < n) dst[id] = (bf16_t)src[id];
}

// ---------------- padded cast: [rows,K] f32 -> [rows,KP] bf16 (zero pad) ----------------
__global__ void padcast_kernel(const float* __restrict__ src, bf16_t* __restrict__ dst,
                               int rows, int K, int KP)
{
    size_t id = (size_t)blockIdx.x * blockDim.x + threadIdx.x;
    const size_t total = (size_t)rows * KP;
    if (id >= total) return;
    int k = (int)(id % KP);
    int r = (int)(id / KP);
    dst[id] = (k < K) ? (bf16_t)src[(size_t)r * K + k] : (bf16_t)0.f;
}

// ---------------- SwiGLU (buggy swish, faithful to reference) ----------------
__global__ void swiglu_kernel(const float* __restrict__ h1, const float* __restrict__ h2,
                              bf16_t* __restrict__ hb)
{
    size_t id = (size_t)blockIdx.x * blockDim.x + threadIdx.x;
    const size_t total = (size_t)MM * HIDP;
    if (id >= total) return;
    int    k = (int)(id % HIDP);
    size_t m = id / HIDP;
    float v = 0.f;
    if (k < HID) {
        float a = h1[m * HID + k];
        float g = h2[m * HID + k];
        v = a * (1.f + __expf(-a)) * g;   // x * (1 + exp(-x))  [source bug]
    }
    hb[id] = (bf16_t)v;
}

// ---------------- host orchestration ----------------
static inline size_t align256(size_t x) { return (x + 255) & ~(size_t)255; }

extern "C" void kernel_launch(void* const* d_in, const int* in_sizes, int n_in,
                              void* d_out, int out_size, void* d_ws, size_t ws_size,
                              hipStream_t stream)
{
    (void)in_sizes; (void)n_in; (void)out_size; (void)ws_size;
    const float* x     = (const float*)d_in[0];
    const float* ln1_w = (const float*)d_in[1];
    const float* Wq    = (const float*)d_in[2];
    const float* Wk    = (const float*)d_in[3];
    const float* Wv    = (const float*)d_in[4];
    const float* Wo    = (const float*)d_in[5];
    const float* bo    = (const float*)d_in[6];
    const float* w1    = (const float*)d_in[7];
    const float* b1    = (const float*)d_in[8];
    const float* w2    = (const float*)d_in[9];
    const float* b2    = (const float*)d_in[10];
    const float* w3    = (const float*)d_in[11];
    const float* b3    = (const float*)d_in[12];
    const float* ln2_w = (const float*)d_in[13];
    const float* cosT  = (const float*)d_in[14];
    const float* sinT  = (const float*)d_in[15];
    float* out = (float*)d_out;

    // workspace carve-up
    char* p = (char*)d_ws;
    auto alloc = [&](size_t bytes) -> void* { void* r = (void*)p; p += align256(bytes); return r; };
    const size_t MC4 = (size_t)MM * CC * 4, MC2 = (size_t)MM * CC * 2;

    float*  x1f   = (float*) alloc(MC4);
    bf16_t* x1b   = (bf16_t*)alloc(MC2);
    float*  qf    = (float*) alloc(MC4);   // q,k,v contiguous (reused for h1 later)
    float*  kf    = (float*) alloc(MC4);
    float*  vf    = (float*) alloc(MC4);
    bf16_t* qb    = (bf16_t*)alloc(MC2);
    bf16_t* kb    = (bf16_t*)alloc(MC2);
    bf16_t* vtb   = (bf16_t*)alloc(MC2);
    float*  attnf = (float*) alloc(MC4);
    bf16_t* attnb = (bf16_t*)alloc(MC2);
    float*  x2f   = (float*) alloc(MC4);
    float*  x3f   = (float*) alloc(MC4);
    bf16_t* x3b   = (bf16_t*)alloc(MC2);
    float*  h2f   = (float*) alloc((size_t)MM * HID * 4);
    bf16_t* hb    = (bf16_t*)alloc((size_t)MM * HIDP * 2);
    bf16_t* Wqb   = (bf16_t*)alloc((size_t)CC * CC * 2);
    bf16_t* Wkb   = (bf16_t*)alloc((size_t)CC * CC * 2);
    bf16_t* Wvb   = (bf16_t*)alloc((size_t)CC * CC * 2);
    bf16_t* Wob   = (bf16_t*)alloc((size_t)CC * CC * 2);
    bf16_t* w1b   = (bf16_t*)alloc((size_t)HID * CC * 2);
    bf16_t* w2b   = (bf16_t*)alloc((size_t)HID * CC * 2);
    bf16_t* w3b   = (bf16_t*)alloc((size_t)CC * HIDP * 2);
    float*  h1f   = qf;  // alias: q/k/v f32 (48MB) dead before FFN; h1 needs 44.7MB

    auto nblks = [](size_t n) { return (unsigned)((n + 255) / 256); };

    // 1) weights -> bf16
    cast_kernel<<<nblks((size_t)CC * CC), 256, 0, stream>>>(Wq, Wqb, (size_t)CC * CC);
    cast_kernel<<<nblks((size_t)CC * CC), 256, 0, stream>>>(Wk, Wkb, (size_t)CC * CC);
    cast_kernel<<<nblks((size_t)CC * CC), 256, 0, stream>>>(Wv, Wvb, (size_t)CC * CC);
    cast_kernel<<<nblks((size_t)CC * CC), 256, 0, stream>>>(Wo, Wob, (size_t)CC * CC);
    cast_kernel<<<nblks((size_t)HID * CC), 256, 0, stream>>>(w1, w1b, (size_t)HID * CC);
    cast_kernel<<<nblks((size_t)HID * CC), 256, 0, stream>>>(w2, w2b, (size_t)HID * CC);
    padcast_kernel<<<nblks((size_t)CC * HIDP), 256, 0, stream>>>(w3, w3b, CC, HID, HIDP);

    // 2) x1 = rmsnorm(x, ln1_w)
    rmsnorm_kernel<<<MM, 256, 0, stream>>>(x, ln1_w, x1f, x1b);

    // 3) q,k,v projections (WMMA GEMMs, [4096,1024] x [1024,1024]^T)
    dim3 gC(MM / 32, CC / 64);
    wmma_gemm_kernel<<<gC, 32, 0, stream>>>(x1b, Wqb, nullptr, nullptr, qf, MM, CC, CC);
    wmma_gemm_kernel<<<gC, 32, 0, stream>>>(x1b, Wkb, nullptr, nullptr, kf, MM, CC, CC);
    wmma_gemm_kernel<<<gC, 32, 0, stream>>>(x1b, Wvb, nullptr, nullptr, vf, MM, CC, CC);

    // 4) RoPE(q,k) -> bf16 ; V -> [B,H,HS,T] bf16
    rope_kernel<<<nblks((size_t)MM * CC / 2), 256, 0, stream>>>(qf, kf, cosT, sinT, qb, kb);
    vtrans_kernel<<<nblks((size_t)MM * CC), 256, 0, stream>>>(vf, vtb);

    // 5) flash attention
    dim3 gA(TT / 16, HH, BB);
    flash_attn_kernel<<<gA, 32, 0, stream>>>(qb, kb, vtb, attnf);

    // 6) x2 = x1 + attn @ Wo^T + bo
    cast_kernel<<<nblks((size_t)MM * CC), 256, 0, stream>>>(attnf, attnb, (size_t)MM * CC);
    wmma_gemm_kernel<<<gC, 32, 0, stream>>>(attnb, Wob, bo, x1f, x2f, MM, CC, CC);

    // 7) x3 = rmsnorm(x2, ln2_w)
    rmsnorm_kernel<<<MM, 256, 0, stream>>>(x2f, ln2_w, x3f, x3b);

    // 8) FFN: h = swish_bug(x3 w1^T + b1) * (x3 w2^T + b2);  out = x3 + h w3^T + b3
    dim3 gH(MM / 32, (HID + 63) / 64);
    wmma_gemm_kernel<<<gH, 32, 0, stream>>>(x3b, w1b, b1, nullptr, h1f, MM, HID, CC);
    wmma_gemm_kernel<<<gH, 32, 0, stream>>>(x3b, w2b, b2, nullptr, h2f, MM, HID, CC);
    swiglu_kernel<<<nblks((size_t)MM * HIDP), 256, 0, stream>>>(h1f, h2f, hb);
    wmma_gemm_kernel<<<gC, 32, 0, stream>>>(hb, w3b, b3, x3f, out, MM, CC, HIDP);
}
